// LiMNet_49297634623719
// MI455X (gfx1250) — matrix operand order repaired and checked by reading, compiled, not run on hardware
//
#include <hip/hip_runtime.h>
#include <hip/hip_bf16.h>
#include <math.h>

// Problem constants (from reference setup_inputs)
constexpr int EDIM = 128;     // E
constexpr int BB   = 4;       // batch
constexpr int TT   = 512;     // steps
constexpr int NN   = 100000;  // NU == NI
constexpr int K2   = 256;     // 2E (GEMM K)
constexpr int NCOMB = 768;    // 3E * 2 cells (GEMM N)
constexpr int NTILES = NCOMB / 16;  // 48
constexpr int KSL  = K2 / 32;       // 8 k-slices of 32
constexpr int THREADS = 512;        // 16 wave32s

typedef __attribute__((ext_vector_type(16))) __bf16 v16bf;
typedef __attribute__((ext_vector_type(8)))  float  v8f;

// ---------------- async helpers (gfx1250 ISA paths) ----------------
__device__ inline void async_gather_b32(unsigned lds_off, const float* gptr) {
  unsigned long long ga = (unsigned long long)gptr;
  // GLOBAL_LOAD_ASYNC_TO_LDS_B32: per-lane dword copy from global memory to the LDS address in VDST; tracked by ASYNCcnt
  asm volatile("global_load_async_to_lds_b32 %0, %1, off"
               :: "v"(lds_off), "v"(ga) : "memory");
}
__device__ inline void wait_async0() {
  asm volatile("s_wait_asynccnt 0" ::: "memory");
}

// ---------------- weight pre-pack: f32 -> bf16 WMMA-B fragments ----------------
// Combined B (K=256 x N=768): cols [0,384) = w_ih_user^T (A k-order = [ue,ie] = xu),
// cols [384,768) = w_ih_item^T with K halves swapped (xi = [ie,ue]).
// Fragment (n,ksl): 32 lanes x 16 halves; lane = (k_local>=16)*16 + ncol, half = k_local%16.
__global__ void pack_weights(const float* __restrict__ wu,
                             const float* __restrict__ wi,
                             __bf16* __restrict__ wB) {
  int id = blockIdx.x * blockDim.x + threadIdx.x;
  if (id >= NTILES * KSL * 512) return;
  int f    = id >> 9;            // fragment index = n*KSL + ksl
  int r    = id & 511;
  int lane = r >> 4, half = r & 15;
  int n    = f / KSL, ksl = f % KSL;
  int ncol = n * 16 + (lane & 15);
  int kk   = ksl * 32 + ((lane >> 4) << 4) + half;
  float v;
  if (ncol < 384) v = wu[ncol * K2 + kk];
  else            v = wi[(ncol - 384) * K2 + ((kk + 128) & 255)];
  wB[id] = (__bf16)v;
}

// ---------------- persistent sequential-scan kernel (1 workgroup) ----------------
__global__ __launch_bounds__(THREADS, 1)
void limnet_scan(float* __restrict__ umem, float* __restrict__ imem,
                 const int* __restrict__ users, const int* __restrict__ items,
                 const __bf16* __restrict__ wB,
                 const float* __restrict__ b_ih_u, const float* __restrict__ b_hh_u,
                 const float* __restrict__ b_ih_i, const float* __restrict__ b_hh_i,
                 float* __restrict__ out) {
  __shared__ __bf16 sA[KSL * 32 * 16];          // A fragments, 16x256 bf16 (8 KB)
  __shared__ float  sGi[NCOMB * BB];            // GEMM result [col][b] (12 KB)
  __shared__ float  sX[2][2][BB][EDIM];         // async-gather staging f32 (8 KB)
  __shared__ float  sNew[2][BB][EDIM];          // last step outputs (4 KB)
  __shared__ float  sBih[2][384], sBhh[2][384]; // biases (6 KB)

  const int tid  = threadIdx.x;
  const int lane = tid & 31;
  const int wid  = tid >> 5;     // 0..15
  const int b    = tid >> 7;     // 0..3
  const int e    = tid & 127;    // 0..127

  // ---- prologue: biases to LDS, zero A staging (rows 4..15 stay zero forever) ----
  if (tid < 384) {
    sBih[0][tid] = b_ih_u[tid]; sBhh[0][tid] = b_hh_u[tid];
    sBih[1][tid] = b_ih_i[tid]; sBhh[1][tid] = b_hh_i[tid];
  }
  for (int i = tid; i < KSL * 32 * 16; i += THREADS) sA[i] = (__bf16)0.0f;

  // ---- initial async gather for t = 0 ----
  {
    int u0 = users[b], i0 = items[b];
    async_gather_b32((unsigned)(size_t)&sX[0][0][b][e], umem + ((size_t)b * NN + u0) * EDIM + e);
    async_gather_b32((unsigned)(size_t)&sX[0][1][b][e], imem + ((size_t)b * NN + i0) * EDIM + e);
  }
  wait_async0();
  __syncthreads();

  for (int t = 0; t < TT; ++t) {
    const int buf    = t & 1;
    const int u_cur  = users[t * BB + b];
    const int i_cur  = items[t * BB + b];
    const int u_prev = (t > 0) ? users[(t - 1) * BB + b] : -1;
    const int i_prev = (t > 0) ? items[(t - 1) * BB + b] : -1;

    // 1) patch speculative gather against step t-1 scatter; convert -> A fragments
    #pragma unroll
    for (int c = 0; c < 2; ++c) {
      const int idx_cur  = c ? i_cur  : u_cur;
      const int idx_prev = c ? i_prev : u_prev;
      float v = (idx_cur == idx_prev) ? sNew[c][b][e] : sX[buf][c][b][e];
      int kk = c * 128 + e;                       // A k-order: [ue, ie]
      int ksl = kk >> 5, p = kk & 31;             // contiguous A k order (k = 16*a + half):
      int lane_t = ((p >> 4) << 4) + b;           // lane = a*16 + M, a = p[4]
      int half   = p & 15;                        // half = p[3:0]
      sA[(ksl * 32 + lane_t) * 16 + half] = (__bf16)v;
    }

    // 2) speculative async gather for step t+1 into the other buffer
    if (t + 1 < TT) {
      int un = users[(t + 1) * BB + b];
      int in_ = items[(t + 1) * BB + b];
      async_gather_b32((unsigned)(size_t)&sX[buf ^ 1][0][b][e],
                       umem + ((size_t)b * NN + un) * EDIM + e);
      async_gather_b32((unsigned)(size_t)&sX[buf ^ 1][1][b][e],
                       imem + ((size_t)b * NN + in_) * EDIM + e);
    }
    __syncthreads();

    // 3) fused GEMM: D(16x768) = A(16x256) x B(256x768), 3 N-tiles per wave
    v8f acc[3] = { {0,0,0,0,0,0,0,0}, {0,0,0,0,0,0,0,0}, {0,0,0,0,0,0,0,0} };
    #pragma unroll
    for (int ksl = 0; ksl < KSL; ++ksl) {
      v16bf afrag = *(const v16bf*)&sA[(ksl * 32 + lane) * 16];
      #pragma unroll
      for (int j = 0; j < 3; ++j) {
        int n = wid + 16 * j;
        v16bf bfrag = *(const v16bf*)&wB[(size_t)(n * KSL + ksl) * 512 + lane * 16];
        acc[j] = __builtin_amdgcn_wmma_f32_16x16x32_bf16(
            false, afrag, false, bfrag, (short)0, acc[j], false, false);
      }
    }
    // D layout: lanes 0-15 hold N=lane, components 0..3 = batches 0..3
    if (lane < 16) {
      #pragma unroll
      for (int j = 0; j < 3; ++j) {
        int col = (wid + 16 * j) * 16 + lane;
        float4 v4 = make_float4(acc[j][0], acc[j][1], acc[j][2], acc[j][3]);
        *(float4*)&sGi[col * BB] = v4;
      }
    }
    __syncthreads();

    // 4) GRU activation (h=0), output store, scatter, keep output for patching
    #pragma unroll
    for (int c = 0; c < 2; ++c) {
      int base = c * 384;
      float ir = sGi[(base + e) * BB + b]        + sBih[c][e]        + sBhh[c][e];
      float iz = sGi[(base + 128 + e) * BB + b]  + sBih[c][128 + e]  + sBhh[c][128 + e];
      float in_ = sGi[(base + 256 + e) * BB + b] + sBih[c][256 + e];
      float hn = sBhh[c][256 + e];
      float rg = 1.0f / (1.0f + __expf(-ir));
      float zg = 1.0f / (1.0f + __expf(-iz));
      float ng = tanhf(in_ + rg * hn);
      float o  = (1.0f - zg) * ng;
      sNew[c][b][e] = o;
      out[((size_t)c * TT + t) * (BB * EDIM) + b * EDIM + e] = o;   // (2,T,B,E)
      int idx = c ? i_cur : u_cur;
      float* mem = c ? imem : umem;
      mem[((size_t)b * NN + idx) * EDIM + e] = o;                   // scatter
    }
    __threadfence();   // scatter(t) visible before gathers issued for t+2
    wait_async0();     // staging for t+1 landed in LDS
    __syncthreads();
  }
}

extern "C" void kernel_launch(void* const* d_in, const int* in_sizes, int n_in,
                              void* d_out, int out_size, void* d_ws, size_t ws_size,
                              hipStream_t stream) {
  const float* user_memory = (const float*)d_in[0];
  const float* item_memory = (const float*)d_in[1];
  const int*   users       = (const int*)d_in[2];
  const int*   items       = (const int*)d_in[3];
  const float* w_ih_user   = (const float*)d_in[4];
  // d_in[5] = w_hh_user: unused (h == 0)
  const float* b_ih_user   = (const float*)d_in[6];
  const float* b_hh_user   = (const float*)d_in[7];
  const float* w_ih_item   = (const float*)d_in[8];
  // d_in[9] = w_hh_item: unused
  const float* b_ih_item   = (const float*)d_in[10];
  const float* b_hh_item   = (const float*)d_in[11];

  const size_t memBytes = (size_t)BB * NN * EDIM * sizeof(float); // 204.8 MB each
  char*   ws   = (char*)d_ws;
  float*  umem = (float*)ws;
  float*  imem = (float*)(ws + memBytes);
  __bf16* wB   = (__bf16*)(ws + 2 * memBytes);

  // Inputs must not be mutated: refresh working copies every launch (graph-safe d2d).
  hipMemcpyAsync(umem, user_memory, memBytes, hipMemcpyDeviceToDevice, stream);
  hipMemcpyAsync(imem, item_memory, memBytes, hipMemcpyDeviceToDevice, stream);

  const int packN = NTILES * KSL * 512;
  pack_weights<<<(packN + 255) / 256, 256, 0, stream>>>(w_ih_user, w_ih_item, wB);

  limnet_scan<<<1, THREADS, 0, stream>>>(umem, imem, users, items, wB,
                                         b_ih_user, b_hh_user, b_ih_item, b_hh_item,
                                         (float*)d_out);
}
